// RNN_52355651338949
// MI455X (gfx1250) — matrix operand order repaired and checked by reading, not compiled, not run on hardware
//
#include <hip/hip_runtime.h>
#include <math.h>

// RNN: h_t = tanh(W x_t + U h_{t-1} + b),  T=2048, I=128, H=256, B=256.
// Batch columns are independent recurrences -> 16 workgroups x 16 columns,
// each iterating all T steps with only workgroup barriers. bf16 WMMA, f32 accum.
// A-fragments of U and W persist in VGPRs; h state double-buffered in LDS.

#define T_STEPS 2048
#define I_SIZE  128
#define H_SIZE  256
#define B_SIZE  256

#define HPAD (H_SIZE + 8)   // padded k-stride (ushorts): 132 dwords == 4 mod 64 banks
#define IPAD (I_SIZE + 8)   // 68 dwords == 4 mod 64 banks

typedef __attribute__((ext_vector_type(16))) __bf16 v16bf;
typedef __attribute__((ext_vector_type(8)))  float  v8f;

union FragBF {
    v16bf v;
    unsigned int u[8];
    uint4 q[2];
};

__device__ __forceinline__ unsigned short f2bf(float f) {
    unsigned int u = __float_as_uint(f);
    u += 0x7FFFu + ((u >> 16) & 1u);          // round-to-nearest-even
    return (unsigned short)(u >> 16);
}

#if __has_builtin(__builtin_amdgcn_cvt_pk_bf16_f32)
typedef __attribute__((ext_vector_type(2))) __bf16 v2bf;
__device__ __forceinline__ unsigned int packbf(float lo, float hi) {
    v2bf p = __builtin_amdgcn_cvt_pk_bf16_f32(lo, hi);
    return *(unsigned int*)&p;
}
#else
__device__ __forceinline__ unsigned int packbf(float lo, float hi) {
    return (unsigned int)f2bf(lo) | ((unsigned int)f2bf(hi) << 16);
}
#endif

#if __has_builtin(__builtin_amdgcn_tanhf)
#define DEV_TANH(x) __builtin_amdgcn_tanhf(x)
#elif __has_builtin(__builtin_amdgcn_tanh_f32)
#define DEV_TANH(x) __builtin_amdgcn_tanh_f32(x)
#else
#define DEV_TANH(x) tanhf(x)
#endif

__global__ __launch_bounds__(512, 1)
void rnn_fused_kernel(const float* __restrict__ x,
                      const float* __restrict__ W,
                      const float* __restrict__ U,
                      const float* __restrict__ b_ih,
                      const float* __restrict__ b_hh,
                      float* __restrict__ out)
{
    // h state and staged x_t slice, column-major bf16: [n][k], double buffered.
    __shared__ __align__(16) unsigned short h_lds[2][16][HPAD];
    __shared__ __align__(16) unsigned short x_lds[2][16][IPAD];

    const int tid  = threadIdx.x;
    const int wave = tid >> 5;
    const int lane = tid & 31;
    const int n    = lane & 15;   // N index (batch col within slice) / A-row select
    const int hh   = lane >> 4;   // lane half
    const int m0   = wave << 4;   // this wave's 16 output rows
    const int col0 = blockIdx.x << 4;   // this WG's 16 batch columns

    // ---- persistent A fragments in registers (loaded once) ----
    // A 16x32 bf16 layout: lane m = (lane&15); VGPR v<4 -> k = hh*8+2v+d,
    // v>=4 -> k = 16+hh*8+2(v-4)+d  => two runs of 8 consecutive K per lane.
    FragBF uf[8];   // U rows m0..m0+15, K = 0..255 in 8 chunks of 32
    FragBF wf[4];   // W rows m0..m0+15, K = 0..127 in 4 chunks of 32
    {
        const int mA = m0 + n;
        const float* urow = U + (size_t)mA * H_SIZE;
#pragma unroll
        for (int c = 0; c < 8; ++c) {
            const int k0 = c * 32 + hh * 8;
            const float4* p0 = (const float4*)(urow + k0);
            const float4* p1 = (const float4*)(urow + k0 + 16);
            float4 a = p0[0], b = p0[1], e = p1[0], f = p1[1];
            uf[c].u[0] = packbf(a.x, a.y); uf[c].u[1] = packbf(a.z, a.w);
            uf[c].u[2] = packbf(b.x, b.y); uf[c].u[3] = packbf(b.z, b.w);
            uf[c].u[4] = packbf(e.x, e.y); uf[c].u[5] = packbf(e.z, e.w);
            uf[c].u[6] = packbf(f.x, f.y); uf[c].u[7] = packbf(f.z, f.w);
        }
        const float* wrow = W + (size_t)mA * I_SIZE;
#pragma unroll
        for (int c = 0; c < 4; ++c) {
            const int k0 = c * 32 + hh * 8;
            const float4* p0 = (const float4*)(wrow + k0);
            const float4* p1 = (const float4*)(wrow + k0 + 16);
            float4 a = p0[0], b = p0[1], e = p1[0], f = p1[1];
            wf[c].u[0] = packbf(a.x, a.y); wf[c].u[1] = packbf(a.z, a.w);
            wf[c].u[2] = packbf(b.x, b.y); wf[c].u[3] = packbf(b.z, b.w);
            wf[c].u[4] = packbf(e.x, e.y); wf[c].u[5] = packbf(e.z, e.w);
            wf[c].u[6] = packbf(f.x, f.y); wf[c].u[7] = packbf(f.z, f.w);
        }
    }

    // ---- fused bias per C/D slot: D row m = m0 + 8*hh + v ----
    float bias_r[8];
#pragma unroll
    for (int v = 0; v < 8; ++v) {
        const int m = m0 + hh * 8 + v;
        bias_r[v] = b_ih[m] + b_hh[m];
    }

    // ---- h_0 = 0 ; stage x_0 ----
    for (int e = tid; e < 16 * HPAD; e += 512)
        ((unsigned short*)h_lds[0])[e] = 0;
    {
#pragma unroll
        for (int j = 0; j < 4; ++j) {
            const int e = tid + j * 512;       // e = i*16 + b  (coalesced over b)
            const int i = e >> 4, b = e & 15;
            float v = x[(size_t)i * B_SIZE + col0 + b];
            x_lds[0][b][i] = f2bf(v);
        }
    }
    __syncthreads();

    // One timestep: read buffers[src], write buffers[dst]. src/dst are
    // compile-time constants at each call site (loop unrolled by 2), so all
    // LDS addresses fold to literal offsets.
    auto step = [&](int t, int src, int dst) {
        // prefetch x_{t+1} into registers (hidden under WMMA work)
        float xr[4];
        if (t + 1 < T_STEPS) {
            const float* xt = x + (size_t)(t + 1) * I_SIZE * B_SIZE;
#pragma unroll
            for (int j = 0; j < 4; ++j) {
                const int e = tid + j * 512;
                const int i = e >> 4, b = e & 15;
                xr[j] = xt[(size_t)i * B_SIZE + col0 + b];
            }
        }

        // Two independent accumulation chains (U-chain and W-chain) so the
        // matrix pipe can overlap them within the wave.
        v8f accU = (v8f)(0.0f);
        v8f accW = (v8f)(0.0f);

        // U @ h_{t-1}: 8 K-chunks of 32. B frag = two runs of 8 bf16 per lane (k=hh*8.., 16+hh*8..).
#pragma unroll
        for (int c = 0; c < 8; ++c) {
            FragBF bf;
            const uint4* p = (const uint4*)&h_lds[src][n][c * 32 + hh * 8];
            bf.q[0] = p[0];
            bf.q[1] = p[2];
            accU = __builtin_amdgcn_wmma_f32_16x16x32_bf16(
                false, uf[c].v, false, bf.v, (short)0, accU, false, false);
        }
        // W @ x_t: 4 K-chunks of 32
#pragma unroll
        for (int c = 0; c < 4; ++c) {
            FragBF bf;
            const uint4* p = (const uint4*)&x_lds[src][n][c * 32 + hh * 8];
            bf.q[0] = p[0];
            bf.q[1] = p[2];
            accW = __builtin_amdgcn_wmma_f32_16x16x32_bf16(
                false, wf[c].v, false, bf.v, (short)0, accW, false, false);
        }

        // activation
        float hv[8];
#pragma unroll
        for (int v = 0; v < 8; ++v)
            hv[v] = DEV_TANH((accU[v] + accW[v]) + bias_r[v]);

        // global store: out[t][m][col0+n], 16 lanes -> 64B contiguous per row
        {
            float* orow = out + (size_t)t * H_SIZE * B_SIZE
                              + (size_t)(m0 + hh * 8) * B_SIZE + col0 + n;
#pragma unroll
            for (int v = 0; v < 8; ++v) orow[(size_t)v * B_SIZE] = hv[v];
            if (t == T_STEPS - 1) {   // h_T appended after outputs
                float* hT = out + (size_t)T_STEPS * H_SIZE * B_SIZE
                                + (size_t)(m0 + hh * 8) * B_SIZE + col0 + n;
#pragma unroll
                for (int v = 0; v < 8; ++v) hT[(size_t)v * B_SIZE] = hv[v];
            }
        }

        // write h_t (bf16) into the other buffer: 8 consecutive k's -> one b128
        {
            uint4 pk;
            pk.x = packbf(hv[0], hv[1]);
            pk.y = packbf(hv[2], hv[3]);
            pk.z = packbf(hv[4], hv[5]);
            pk.w = packbf(hv[6], hv[7]);
            *(uint4*)&h_lds[dst][n][m0 + hh * 8] = pk;
        }
        // stage x_{t+1}
        if (t + 1 < T_STEPS) {
#pragma unroll
            for (int j = 0; j < 4; ++j) {
                const int e = tid + j * 512;
                const int i = e >> 4, b = e & 15;
                x_lds[dst][b][i] = f2bf(xr[j]);
            }
        }

        __syncthreads();   // one barrier per timestep
    };

#pragma unroll 1
    for (int t = 0; t < T_STEPS; t += 2) {
        step(t,     0, 1);
        step(t + 1, 1, 0);
    }
}

extern "C" void kernel_launch(void* const* d_in, const int* in_sizes, int n_in,
                              void* d_out, int out_size, void* d_ws, size_t ws_size,
                              hipStream_t stream)
{
    (void)in_sizes; (void)n_in; (void)d_ws; (void)ws_size; (void)out_size;
    const float* x    = (const float*)d_in[0];
    const float* W    = (const float*)d_in[1];
    const float* U    = (const float*)d_in[2];
    const float* b_ih = (const float*)d_in[3];
    const float* b_hh = (const float*)d_in[4];
    float* out = (float*)d_out;

    rnn_fused_kernel<<<dim3(16), dim3(512), 0, stream>>>(x, W, U, b_ih, b_hh, out);
}
